// DAROD_30743375905168
// MI455X (gfx1250) — compile-verified, matrix operand order checked by reading, not run on hardware
//
#include <hip/hip_runtime.h>
#include <hip/hip_bf16.h>
#include <math.h>

// ---------------------------------------------------------------------------
// Types / helpers
// ---------------------------------------------------------------------------
typedef __bf16 bf16_t;
typedef __bf16 v16bf __attribute__((ext_vector_type(16)));
typedef float  v8f   __attribute__((ext_vector_type(8)));

__device__ __forceinline__ float bf2f(bf16_t b) {
    unsigned short h = __builtin_bit_cast(unsigned short, b);
    unsigned u = ((unsigned)h) << 16;
    return __builtin_bit_cast(float, u);
}
__device__ __forceinline__ bf16_t f2bf(float f) {
    unsigned u = __builtin_bit_cast(unsigned, f);
    u += 0x7fffu + ((u >> 16) & 1u);   // round-to-nearest-even
    return __builtin_bit_cast(bf16_t, (unsigned short)(u >> 16));
}

#define NEG_BIG (-3.402823466e38f)

// Fragment = 32 bytes: two b128 chunks.  Memory order of a 32-k row is exactly
// the ISA 7.12.2 16-bit A/B lane layout: lane (g,l15) takes halves
// [g*8 .. g*8+7] and [16+g*8 .. 16+g*8+7] -> uint4 p[g] and p[2+g].
struct Frag32 { uint4 a, b; };
__device__ __forceinline__ v16bf make_frag(const uint4* p, int g) {
    Frag32 u;
    u.a = p[g];
    u.b = p[2 + g];
    return __builtin_bit_cast(v16bf, u);
}

// Bitonic sort, descending by (key desc, idx asc).  n = power of two.
__device__ void bitonic_desc(float* key, unsigned short* idx, int n, int tid, int nth) {
    for (int k = 2; k <= n; k <<= 1) {
        for (int j = k >> 1; j > 0; j >>= 1) {
            for (int t = tid; t < n; t += nth) {
                int ixj = t ^ j;
                if (ixj > t) {
                    float ka = key[t], kb = key[ixj];
                    unsigned short ia = idx[t], ib = idx[ixj];
                    bool aBetter = (ka > kb) || (ka == kb && ia < ib);
                    bool up = ((t & k) == 0);
                    if (up ? !aBetter : aBetter) {
                        key[t] = kb; key[ixj] = ka;
                        idx[t] = ib; idx[ixj] = ia;
                    }
                }
            }
            __syncthreads();
        }
    }
}

__device__ __forceinline__ void apply_deltas4(const float a[4], const float d[4], float o[4]) {
    float ah = a[2] - a[0], aw = a[3] - a[1];
    float acy = a[0] + 0.5f * ah, acx = a[1] + 0.5f * aw;
    float h = expf(d[2]) * ah;
    float w = expf(d[3]) * aw;
    float cy = d[0] * ah + acy, cx = d[1] * aw + acx;
    o[0] = cy - 0.5f * h; o[1] = cx - 0.5f * w;
    o[2] = cy + 0.5f * h; o[3] = cx + 0.5f * w;
}

// ---------------------------------------------------------------------------
// Weight conversion into fragment-ready blocked layout dst[kb][n][32] (bf16).
// FC:   k = kb*32+kk over [0,Ksrc) padded to Kpad, src is (N x Ksrc) row-major.
// Conv: k = t*Cin + ci  (tap-major, channel-minor), src is OIHW (N x Cin x 3 x 3).
// ---------------------------------------------------------------------------
__global__ void convertB_fc_kernel(const float* __restrict__ src, bf16_t* __restrict__ dst,
                                   int Ksrc, int Kpad, int N) {
    int gid = blockIdx.x * blockDim.x + threadIdx.x;
    if (gid >= Kpad * N) return;
    int kk = gid & 31;
    int tmp = gid >> 5;
    int n = tmp % N;
    int kb = tmp / N;
    int k = (kb << 5) + kk;
    float v = (k < Ksrc) ? src[(size_t)n * Ksrc + k] : 0.f;
    dst[gid] = f2bf(v);
}

__global__ void convertB_conv_kernel(const float* __restrict__ src, bf16_t* __restrict__ dst,
                                     int Cin, int N) {
    int gid = blockIdx.x * blockDim.x + threadIdx.x;
    if (gid >= Cin * 9 * N) return;
    int kk = gid & 31;
    int tmp = gid >> 5;
    int n = tmp % N;
    int kb = tmp / N;
    int k = (kb << 5) + kk;          // k = t*Cin + ci
    int t = k / Cin;
    int ci = k - t * Cin;
    float v = src[(size_t)n * Cin * 9 + ci * 9 + t];
    dst[gid] = f2bf(v);
}

// ---------------------------------------------------------------------------
// First conv: Cin=1, 3x3 SAME, leaky 0.01.  x:(16,1,256,64) fp32 -> NHWC bf16
// ---------------------------------------------------------------------------
__global__ void conv1_kernel(const float* __restrict__ x, const float* __restrict__ w,
                             const float* __restrict__ bias, bf16_t* __restrict__ out) {
    int gid = blockIdx.x * blockDim.x + threadIdx.x;
    if (gid >= (1 << 24)) return;
    int o  = gid & 63;
    int xx = (gid >> 6) & 63;
    int y  = (gid >> 12) & 255;
    int b  = gid >> 20;
    float acc = bias[o];
#pragma unroll
    for (int dy = 0; dy < 3; ++dy)
#pragma unroll
        for (int dx = 0; dx < 3; ++dx) {
            int yy = y + dy - 1, xi = xx + dx - 1;
            if ((unsigned)yy < 256u && (unsigned)xi < 64u)
                acc += x[(b << 14) + (yy << 6) + xi] * w[o * 9 + dy * 3 + dx];
        }
    acc = acc > 0.f ? acc : 0.01f * acc;
    out[gid] = f2bf(acc);          // gid == ((b*256+y)*64+xx)*64 + o
}

// ---------------------------------------------------------------------------
// Implicit-GEMM 3x3 conv, NHWC activations, bf16 WMMA.
// M = B*H*W (%128==0), N = Cout (%64==0), K ordered (dy,dx)-major / ci-minor.
// Block = 256 thr (8 waves): 128 pixels x 64 couts.  Wave: 32x32 via 2x2 wmma.
// B fragments are read directly from blocked weights in global (L2-resident).
// ---------------------------------------------------------------------------
__global__ __launch_bounds__(256) void conv3x3_wmma_kernel(
    const bf16_t* __restrict__ in, const bf16_t* __restrict__ wBblk,
    const float* __restrict__ bias, bf16_t* __restrict__ out,
    int Cin, int H, int W, int Cout, int relu_mode) {
    const int HW = H * W;
    const int M0 = blockIdx.x * 128;
    const int N0 = blockIdx.y * 64;
    __shared__ __align__(16) bf16_t As[128][40];   // 80B rows -> b128-aligned
    const int tid  = threadIdx.x;
    const int lane = tid & 31;
    const int wv   = tid >> 5;
    const int wm = wv >> 1, wn = wv & 1;
    const int g = lane >> 4, l15 = lane & 15;

    // staging prologue: 4 slots of (row, 4-channel chunk); hoist pixel decomp
    int srow[4], sch[4], sm[4], sy[4], sx[4];
#pragma unroll
    for (int r = 0; r < 4; ++r) {
        int idx = tid + r * 256;      // 1024 = 128 rows * 8 chunks
        int row = idx >> 3;
        int ch4 = (idx & 7) << 2;
        int m = M0 + row;
        int b = m / HW; int rem = m - b * HW;
        srow[r] = row; sch[r] = ch4; sm[r] = m;
        sy[r] = rem / W; sx[r] = rem - sy[r] * W;
    }

    v8f z = {0.f, 0.f, 0.f, 0.f, 0.f, 0.f, 0.f, 0.f};
    v8f acc00 = z, acc01 = z, acc10 = z, acc11 = z;
    const int KB = Cin >> 5;

    for (int t = 0; t < 9; ++t) {
        const int dy = t / 3 - 1, dx = t % 3 - 1;
        const int doff = dy * W + dx;
        for (int cb = 0; cb < KB; ++cb) {
            const int ci0 = cb << 5;
            __syncthreads();
#pragma unroll
            for (int r = 0; r < 4; ++r) {
                uint2 v; v.x = 0u; v.y = 0u;
                int yy = sy[r] + dy, xx = sx[r] + dx;
                if ((unsigned)yy < (unsigned)H && (unsigned)xx < (unsigned)W)
                    v = *(const uint2*)(in + (size_t)(sm[r] + doff) * Cin + ci0 + sch[r]);
                *(uint2*)&As[srow[r]][sch[r]] = v;
            }
            if (cb + 1 < KB)
                __builtin_prefetch(in + (size_t)(sm[0] + doff) * Cin + ci0 + 32 + sch[0], 0, 0);
            __syncthreads();

            const int kb = t * KB + cb;
            const uint4* pb0 = (const uint4*)(wBblk + ((size_t)kb * Cout + N0 + wn * 32 + l15) * 32);
            const uint4* pb1 = (const uint4*)(wBblk + ((size_t)kb * Cout + N0 + wn * 32 + 16 + l15) * 32);
            v16bf b0 = make_frag(pb0, g);
            v16bf b1 = make_frag(pb1, g);
            const uint4* pa0 = (const uint4*)&As[wm * 32 + l15][0];
            const uint4* pa1 = (const uint4*)&As[wm * 32 + 16 + l15][0];
            v16bf a0 = make_frag(pa0, g);
            v16bf a1 = make_frag(pa1, g);
            acc00 = __builtin_amdgcn_wmma_f32_16x16x32_bf16(false, a0, false, b0, (short)0, acc00, false, false);
            acc01 = __builtin_amdgcn_wmma_f32_16x16x32_bf16(false, a0, false, b1, (short)0, acc01, false, false);
            acc10 = __builtin_amdgcn_wmma_f32_16x16x32_bf16(false, a1, false, b0, (short)0, acc10, false, false);
            acc11 = __builtin_amdgcn_wmma_f32_16x16x32_bf16(false, a1, false, b1, (short)0, acc11, false, false);
        }
    }

    // epilogue: NHWC -> out[m*Cout + n]; lane holds fixed n, 8 m's per subtile
#pragma unroll
    for (int i = 0; i < 2; ++i) {
#pragma unroll
        for (int j = 0; j < 2; ++j) {
            v8f acc = (i == 0) ? ((j == 0) ? acc00 : acc01) : ((j == 0) ? acc10 : acc11);
            int n = N0 + wn * 32 + j * 16 + l15;
            float bval = bias[n];
#pragma unroll
            for (int r = 0; r < 8; ++r) {
                int m = M0 + wm * 32 + i * 16 + g * 8 + r;
                float v = acc[r] + bval;
                v = relu_mode ? fmaxf(v, 0.f) : (v > 0.f ? v : 0.01f * v);
                out[(size_t)m * Cout + n] = f2bf(v);
            }
        }
    }
}

// ---------------------------------------------------------------------------
// Plain bf16 WMMA GEMM: C(MxN) = act(A*B + bias).  M%64==0, N%128==0, K%32==0.
// Block: 64 rows x 128 cols; wave 32x32 via 2x2 wmma; B frags direct from global.
// ---------------------------------------------------------------------------
__global__ __launch_bounds__(256) void gemm_bf16_wmma_kernel(
    const bf16_t* __restrict__ A, const bf16_t* __restrict__ wBblk,
    const float* __restrict__ bias, bf16_t* __restrict__ out,
    int M, int N, int K, int relu_mode) {
    const int M0 = blockIdx.x * 64;
    const int N0 = blockIdx.y * 128;
    __shared__ __align__(16) bf16_t As[64][40];
    const int tid  = threadIdx.x;
    const int lane = tid & 31;
    const int wv   = tid >> 5;
    const int wm = wv & 1, wn = wv >> 1;
    const int g = lane >> 4, l15 = lane & 15;
    (void)M;

    int srow[2], sch[2];
#pragma unroll
    for (int r = 0; r < 2; ++r) {
        int idx = tid + r * 256;      // 512 = 64 rows * 8 chunks
        srow[r] = idx >> 3;
        sch[r]  = (idx & 7) << 2;
    }

    v8f z = {0.f, 0.f, 0.f, 0.f, 0.f, 0.f, 0.f, 0.f};
    v8f acc00 = z, acc01 = z, acc10 = z, acc11 = z;
    const int KB = K >> 5;

    for (int kb = 0; kb < KB; ++kb) {
        __syncthreads();
#pragma unroll
        for (int r = 0; r < 2; ++r)
            *(uint2*)&As[srow[r]][sch[r]] =
                *(const uint2*)(A + (size_t)(M0 + srow[r]) * K + (kb << 5) + sch[r]);
        if (kb + 1 < KB)
            __builtin_prefetch(A + (size_t)(M0 + srow[0]) * K + ((kb + 1) << 5) + sch[0], 0, 0);
        __syncthreads();

        const uint4* pb0 = (const uint4*)(wBblk + ((size_t)kb * N + N0 + wn * 32 + l15) * 32);
        const uint4* pb1 = (const uint4*)(wBblk + ((size_t)kb * N + N0 + wn * 32 + 16 + l15) * 32);
        v16bf b0 = make_frag(pb0, g);
        v16bf b1 = make_frag(pb1, g);
        const uint4* pa0 = (const uint4*)&As[wm * 32 + l15][0];
        const uint4* pa1 = (const uint4*)&As[wm * 32 + 16 + l15][0];
        v16bf a0 = make_frag(pa0, g);
        v16bf a1 = make_frag(pa1, g);
        acc00 = __builtin_amdgcn_wmma_f32_16x16x32_bf16(false, a0, false, b0, (short)0, acc00, false, false);
        acc01 = __builtin_amdgcn_wmma_f32_16x16x32_bf16(false, a0, false, b1, (short)0, acc01, false, false);
        acc10 = __builtin_amdgcn_wmma_f32_16x16x32_bf16(false, a1, false, b0, (short)0, acc10, false, false);
        acc11 = __builtin_amdgcn_wmma_f32_16x16x32_bf16(false, a1, false, b1, (short)0, acc11, false, false);
    }

#pragma unroll
    for (int i = 0; i < 2; ++i) {
#pragma unroll
        for (int j = 0; j < 2; ++j) {
            v8f acc = (i == 0) ? ((j == 0) ? acc00 : acc01) : ((j == 0) ? acc10 : acc11);
            int n = N0 + wn * 32 + j * 16 + l15;
            float bval = bias[n];
#pragma unroll
            for (int r = 0; r < 8; ++r) {
                int m = M0 + wm * 32 + i * 16 + g * 8 + r;
                float v = acc[r] + bval;
                if (relu_mode) v = fmaxf(v, 0.f);
                out[(size_t)m * N + n] = f2bf(v);
            }
        }
    }
}

// ---------------------------------------------------------------------------
// Max pool, NHWC (leaky already applied).  Optional fp32 copy (for fm).
// ---------------------------------------------------------------------------
__global__ void maxpool_kernel(const bf16_t* __restrict__ in, bf16_t* __restrict__ outb,
                               float* __restrict__ outf, int C, int H, int W, int ph, int pw) {
    int Ho = H / ph, Wo = W / pw;
    int total = 16 * C * Ho * Wo;
    int gid = blockIdx.x * blockDim.x + threadIdx.x;
    if (gid >= total) return;
    int c = gid % C; int tmp = gid / C;
    int x = tmp % Wo; tmp /= Wo;
    int y = tmp % Ho; int b = tmp / Ho;
    float m = -INFINITY;
    for (int dy = 0; dy < ph; ++dy)
        for (int dx = 0; dx < pw; ++dx) {
            float v = bf2f(in[(size_t)(((b * H + y * ph + dy) * W) + x * pw + dx) * C + c]);
            m = fmaxf(m, v);
        }
    outb[gid] = f2bf(m);
    if (outf) outf[gid] = m;
}

// ---------------------------------------------------------------------------
// RPN 1x1 heads: act NHWC (16,32,32,256) -> rpn_cls NHWC(...,5), rpn_reg NHWC(...,20)
// ---------------------------------------------------------------------------
__global__ __launch_bounds__(256) void rpn_heads_kernel(
    const bf16_t* __restrict__ act, const float* __restrict__ wc, const float* __restrict__ bc,
    const float* __restrict__ wr, const float* __restrict__ br,
    float* __restrict__ Ocls, float* __restrict__ Oreg) {
    int b = blockIdx.x >> 10;
    int pix = blockIdx.x & 1023;
    __shared__ float sact[256];
    sact[threadIdx.x] = bf2f(act[(size_t)(b * 1024 + pix) * 256 + threadIdx.x]);
    __syncthreads();
    int t = threadIdx.x;
    if (t < 25) {
        const float* w = (t < 5) ? (wc + t * 256) : (wr + (t - 5) * 256);
        float acc = (t < 5) ? bc[t] : br[t - 5];
        for (int k = 0; k < 256; ++k) acc += sact[k] * w[k];
        if (t < 5) Ocls[(b * 1024 + pix) * 5 + t] = acc;
        else       Oreg[(b * 1024 + pix) * 20 + (t - 5)] = acc;
    }
}

// ---------------------------------------------------------------------------
// RPN decode: sigmoid scores + anchor deltas -> boxes
// ---------------------------------------------------------------------------
__global__ void rpn_decode_kernel(const float* __restrict__ cls_nhwc,
                                  const float* __restrict__ reg_nhwc,
                                  float* __restrict__ scores, float* __restrict__ boxes) {
    const float AW[5] = {0.1f, 0.2f, 0.35f, 0.07071068f, 0.14142136f};
    const float AH[5] = {0.1f, 0.2f, 0.35f, 0.14142136f, 0.07071068f};
    const float VAR[4] = {0.1f, 0.1f, 0.2f, 0.2f};
    int gid = blockIdx.x * blockDim.x + threadIdx.x;
    if (gid >= 16 * 5120) return;
    int b = gid / 5120, i = gid - b * 5120;
    int pos = i / 5, a = i - pos * 5;
    int gy = pos >> 5, gx = pos & 31;
    float sx = (gx + 0.5f) / 32.f, sy = (gy + 0.5f) / 32.f;
    float w2 = 0.5f * AW[a], h2 = 0.5f * AH[a];
    float an[4] = {sx - w2, sy - h2, sx + w2, sy + h2};
    bool bad = false;
    for (int q = 0; q < 4; ++q) bad = bad || (an[q] <= 0.f) || (an[q] >= 1.f);
    if (bad) { an[0] = an[1] = an[2] = an[3] = 0.f; }
    float d[4], o[4];
    for (int q = 0; q < 4; ++q) d[q] = reg_nhwc[b * 20480 + i * 4 + q] * VAR[q];
    apply_deltas4(an, d, o);
    scores[gid] = 1.f / (1.f + expf(-cls_nhwc[gid]));
    for (int q = 0; q < 4; ++q) boxes[gid * 4 + q] = o[q];
}

// ---------------------------------------------------------------------------
// RPN top-1000 per image (bitonic over 8192 in LDS)
// ---------------------------------------------------------------------------
__global__ __launch_bounds__(1024) void rpn_topk_kernel(
    const float* __restrict__ scores, const float* __restrict__ boxes,
    float* __restrict__ tops, float* __restrict__ topb) {
    __shared__ float key[8192];
    __shared__ unsigned short sidx[8192];
    int b = blockIdx.x, tid = threadIdx.x;
    for (int t = tid; t < 8192; t += 1024) {
        key[t] = (t < 5120) ? scores[b * 5120 + t] : NEG_BIG;
        sidx[t] = (unsigned short)t;
    }
    __syncthreads();
    bitonic_desc(key, sidx, 8192, tid, 1024);
    for (int t = tid; t < 1000; t += 1024) {
        tops[b * 1000 + t] = key[t];
        int id = sidx[t];
        for (int q = 0; q < 4; ++q)
            topb[(b * 1000 + t) * 4 + q] = boxes[(b * 5120 + id) * 4 + q];
    }
}

// ---------------------------------------------------------------------------
// RPN greedy NMS (IoU 0.7) + stable compaction to 300 rois
// ---------------------------------------------------------------------------
__global__ __launch_bounds__(1024) void rpn_nms_kernel(
    const float* __restrict__ topb, const float* __restrict__ tops,
    float* __restrict__ roib, float* __restrict__ rois) {
    int b = blockIdx.x, tid = threadIdx.x;
    __shared__ float bx[1000][4];
    __shared__ float area[1000];
    __shared__ unsigned char keep[1000];
    __shared__ int ord[300];
    __shared__ int cnt;
    for (int t = tid; t < 1000; t += 1024) {
        for (int q = 0; q < 4; ++q) bx[t][q] = topb[(b * 1000 + t) * 4 + q];
        area[t] = (bx[t][2] - bx[t][0]) * (bx[t][3] - bx[t][1]);
        keep[t] = 1;
    }
    __syncthreads();
    for (int i = 0; i < 1000; ++i) {
        if (keep[i]) {
            float y1 = bx[i][0], x1 = bx[i][1], y2 = bx[i][2], x2 = bx[i][3], ai = area[i];
            for (int j = tid; j < 1000; j += 1024) {
                if (j > i && keep[j]) {
                    float ih = fmaxf(fminf(y2, bx[j][2]) - fmaxf(y1, bx[j][0]), 0.f);
                    float iw = fmaxf(fminf(x2, bx[j][3]) - fmaxf(x1, bx[j][1]), 0.f);
                    float inter = ih * iw;
                    float un = ai + area[j] - inter;
                    float iou = (un > 0.f) ? inter / un : 0.f;
                    if (iou > 0.7f) keep[j] = 0;
                }
            }
        }
        __syncthreads();
    }
    if (tid == 0) {
        int c = 0;
        for (int j = 0; j < 1000; ++j)
            if (keep[j] && c < 300) ord[c++] = j;
        cnt = c;
    }
    __syncthreads();
    for (int t = tid; t < 300; t += 1024) {
        if (t < cnt) {
            int id = ord[t];
            for (int q = 0; q < 4; ++q) roib[(b * 300 + t) * 4 + q] = bx[id][q];
            rois[b * 300 + t] = tops[b * 1000 + id];
        } else {
            for (int q = 0; q < 4; ++q) roib[(b * 300 + t) * 4 + q] = 0.f;
            rois[b * 300 + t] = 0.f;
        }
    }
}

// ---------------------------------------------------------------------------
// RoI pool (4x4 bilinear, NHWC fm) -> bf16 rows of K=4128
// ---------------------------------------------------------------------------
__global__ __launch_bounds__(256) void roi_pool_kernel(
    const float* __restrict__ fm, const float* __restrict__ roib,
    const float* __restrict__ rois, bf16_t* __restrict__ pooled) {
    int row = blockIdx.x;              // b*300 + n
    int b = row / 300;
    __shared__ int iy0[16], iy1[16], ix0[16], ix1[16];
    __shared__ float wy[16], wx[16];
    __shared__ unsigned char inb[16];
    __shared__ float box[4];
    int tid = threadIdx.x;
    if (tid < 4) box[tid] = roib[row * 4 + tid];
    __syncthreads();
    if (tid < 16) {
        int ph = tid >> 2, pw = tid & 3;
        float ty = ph * (1.f / 3.f), tx = pw * (1.f / 3.f);
        float yy = (box[0] + ty * (box[2] - box[0])) * 31.f;
        float xx = (box[1] + tx * (box[3] - box[1])) * 31.f;
        float y0 = floorf(yy), x0 = floorf(xx);
        wy[tid] = yy - y0; wx[tid] = xx - x0;
        iy0[tid] = (int)fminf(fmaxf(y0, 0.f), 31.f);
        iy1[tid] = (int)fminf(fmaxf(y0 + 1.f, 0.f), 31.f);
        ix0[tid] = (int)fminf(fmaxf(x0, 0.f), 31.f);
        ix1[tid] = (int)fminf(fmaxf(x0 + 1.f, 0.f), 31.f);
        inb[tid] = (yy >= 0.f && yy <= 31.f && xx >= 0.f && xx <= 31.f) ? 1 : 0;
    }
    __syncthreads();
    int c = tid;
    const float* f = fm + (size_t)b * 1024 * 256;
    bf16_t* op = pooled + (size_t)row * 4128 + c * 16;
#pragma unroll
    for (int j = 0; j < 16; ++j) {
        float v = 0.f;
        if (inb[j]) {
            float a  = f[(iy0[j] * 32 + ix0[j]) * 256 + c];
            float b2 = f[(iy0[j] * 32 + ix1[j]) * 256 + c];
            float c2 = f[(iy1[j] * 32 + ix0[j]) * 256 + c];
            float d  = f[(iy1[j] * 32 + ix1[j]) * 256 + c];
            float u = wy[j], t2 = wx[j];
            v = a * (1 - u) * (1 - t2) + b2 * (1 - u) * t2 + c2 * u * (1 - t2) + d * u * t2;
        }
        op[j] = f2bf(v);
    }
    if (tid < 32) {
        float v = 0.f;
        if (tid == 0) {
            float cy = box[0] + 0.5f * (box[2] - box[0]);
            v = (rois[row] > 0.f) ? (1.f - cy) : 0.f;
        } else if (tid == 1) {
            float cx = box[1] + 0.5f * (box[3] - box[1]);
            v = (rois[row] > 0.f) ? (2.f * cx - 1.f) : 0.f;
        }
        pooled[(size_t)row * 4128 + 4096 + tid] = f2bf(v);
    }
}

// ---------------------------------------------------------------------------
// Detector heads: fc2 out (bf16 4800x512) -> cls (4800x4), reg (4800x16) fp32
// ---------------------------------------------------------------------------
__global__ void fc_heads_kernel(const bf16_t* __restrict__ A, const float* __restrict__ wc,
                                const float* __restrict__ bc, const float* __restrict__ wr,
                                const float* __restrict__ br, float* __restrict__ Ocls,
                                float* __restrict__ Oreg) {
    int gid = blockIdx.x * blockDim.x + threadIdx.x;
    if (gid >= 4800 * 20) return;
    int row = gid / 20, j = gid - row * 20;
    const bf16_t* a = A + (size_t)row * 512;
    const float* w;
    float acc;
    if (j < 4) { w = wc + j * 512; acc = bc[j]; }
    else       { w = wr + (j - 4) * 512; acc = br[j - 4]; }
    for (int k = 0; k < 512; ++k) acc += bf2f(a[k]) * w[k];
    if (j < 4) Ocls[row * 4 + j] = acc;
    else       Oreg[row * 16 + (j - 4)] = acc;
}

// ---------------------------------------------------------------------------
// Softmax + argmax over 4 classes
// ---------------------------------------------------------------------------
__global__ void softmax_kernel(const float* __restrict__ cls, float* __restrict__ probs,
                               int* __restrict__ amax) {
    int gid = blockIdx.x * blockDim.x + threadIdx.x;
    if (gid >= 4800) return;
    const float* c = cls + gid * 4;
    int am = 0;
    float m = c[0];
    for (int q = 1; q < 4; ++q) if (c[q] > m) { m = c[q]; am = q; }
    float e[4], s = 0.f;
    for (int q = 0; q < 4; ++q) { e[q] = expf(c[q] - m); s += e[q]; }
    for (int q = 0; q < 4; ++q) probs[gid * 4 + q] = e[q] / s;
    amax[gid] = am;
}

// ---------------------------------------------------------------------------
// Decoder per-(image,class): stable sort by score, class-deltas, greedy NMS,
// score threshold, top-100 cap.  CB: sorted boxes, CS: kept scores.
// ---------------------------------------------------------------------------
__global__ __launch_bounds__(512) void perclass_kernel(
    const float* __restrict__ probs, const int* __restrict__ amax,
    const float* __restrict__ roib, const float* __restrict__ regp,
    float* __restrict__ CBo, float* __restrict__ CSo) {
    const float VAR[4] = {0.1f, 0.1f, 0.2f, 0.2f};
    int blk = blockIdx.x;
    int b = blk >> 2, c = blk & 3;
    int tid = threadIdx.x;
    __shared__ float key[512];
    __shared__ unsigned short sidx[512];
    __shared__ float bs[300][4];
    __shared__ float area[300];
    __shared__ unsigned char keep[300];
    key[tid] = (tid < 300) ? ((amax[b * 300 + tid] != 0) ? probs[(b * 300 + tid) * 4 + c] : 0.f)
                           : NEG_BIG;
    sidx[tid] = (unsigned short)tid;
    __syncthreads();
    bitonic_desc(key, sidx, 512, tid, 512);
    if (tid < 300) {
        int n = sidx[tid];
        float a[4], d[4], o[4];
        for (int q = 0; q < 4; ++q) a[q] = roib[(b * 300 + n) * 4 + q];
        for (int q = 0; q < 4; ++q) d[q] = regp[(b * 300 + n) * 16 + c * 4 + q] * VAR[q];
        apply_deltas4(a, d, o);
        for (int q = 0; q < 4; ++q) bs[tid][q] = o[q];
        area[tid] = (o[2] - o[0]) * (o[3] - o[1]);
        keep[tid] = 1;
    }
    __syncthreads();
    for (int i = 0; i < 300; ++i) {
        if (keep[i]) {
            float y1 = bs[i][0], x1 = bs[i][1], y2 = bs[i][2], x2 = bs[i][3], ai = area[i];
            if (tid < 300 && tid > i && keep[tid]) {
                float ih = fmaxf(fminf(y2, bs[tid][2]) - fmaxf(y1, bs[tid][0]), 0.f);
                float iw = fmaxf(fminf(x2, bs[tid][3]) - fmaxf(x1, bs[tid][1]), 0.f);
                float inter = ih * iw;
                float un = ai + area[tid] - inter;
                float iou = (un > 0.f) ? inter / un : 0.f;
                if (iou > 0.5f) keep[tid] = 0;
            }
        }
        __syncthreads();
    }
    if (tid == 0) {
        int cc = 0;
        for (int j = 0; j < 300; ++j) {
            bool k = keep[j] && (key[j] > 0.5f);
            if (k) { if (cc < 100) cc++; else k = false; }
            keep[j] = k ? 1 : 0;
        }
    }
    __syncthreads();
    if (tid < 300) {
        for (int q = 0; q < 4; ++q) CBo[(blk * 300 + tid) * 4 + q] = bs[tid][q];
        CSo[blk * 300 + tid] = keep[tid] ? key[tid] : 0.f;
    }
}

// ---------------------------------------------------------------------------
// Final top-100 over 1200 (class,box) candidates per image
// ---------------------------------------------------------------------------
__global__ __launch_bounds__(1024) void final_topk_kernel(
    const float* __restrict__ CBo, const float* __restrict__ CSo,
    float* __restrict__ Ob, float* __restrict__ Ol, float* __restrict__ Os) {
    int b = blockIdx.x, tid = threadIdx.x;
    __shared__ float key[2048];
    __shared__ unsigned short sidx[2048];
    for (int t = tid; t < 2048; t += 1024) {
        key[t] = (t < 1200) ? CSo[b * 1200 + t] : NEG_BIG;
        sidx[t] = (unsigned short)t;
    }
    __syncthreads();
    bitonic_desc(key, sidx, 2048, tid, 1024);
    for (int t = tid; t < 100; t += 1024) {
        float v = key[t];
        int f = sidx[t];
        bool ok = v > 0.f;
        int c = f / 300;
        for (int q = 0; q < 4; ++q)
            Ob[(b * 100 + t) * 4 + q] = ok ? CBo[(b * 1200 + f) * 4 + q] : 0.f;
        Ol[b * 100 + t] = ok ? (float)c : 0.f;
        Os[b * 100 + t] = ok ? v : 0.f;
    }
}

// ---------------------------------------------------------------------------
// Host launcher
// ---------------------------------------------------------------------------
extern "C" void kernel_launch(void* const* d_in, const int* in_sizes, int n_in,
                              void* d_out, int out_size, void* d_ws, size_t ws_size,
                              hipStream_t stream) {
    (void)in_sizes; (void)n_in; (void)out_size; (void)ws_size;
    // inputs in setup_inputs() dict insertion order, lists expanded
    const float* x       = (const float*)d_in[0];
    const float* b1w0    = (const float*)d_in[1];
    const float* b1w1    = (const float*)d_in[2];
    const float* b1b0    = (const float*)d_in[3];
    const float* b1b1    = (const float*)d_in[4];
    const float* b2w0    = (const float*)d_in[5];
    const float* b2w1    = (const float*)d_in[6];
    const float* b2b0    = (const float*)d_in[7];
    const float* b2b1    = (const float*)d_in[8];
    const float* b3w0    = (const float*)d_in[9];
    const float* b3w1    = (const float*)d_in[10];
    const float* b3w2    = (const float*)d_in[11];
    const float* b3b0    = (const float*)d_in[12];
    const float* b3b1    = (const float*)d_in[13];
    const float* b3b2    = (const float*)d_in[14];
    const float* rpnw    = (const float*)d_in[15];
    const float* rpnb    = (const float*)d_in[16];
    const float* rpnclsw = (const float*)d_in[17];
    const float* rpnclsb = (const float*)d_in[18];
    const float* rpnregw = (const float*)d_in[19];
    const float* rpnregb = (const float*)d_in[20];
    const float* fc1w    = (const float*)d_in[21];
    const float* fc1b    = (const float*)d_in[22];
    const float* fc2w    = (const float*)d_in[23];
    const float* fc2b    = (const float*)d_in[24];
    const float* clsw    = (const float*)d_in[25];
    const float* clsb    = (const float*)d_in[26];
    const float* regw    = (const float*)d_in[27];
    const float* regb    = (const float*)d_in[28];

    float* out = (float*)d_out;
    float* O1 = out;            // rpn_cls NHWC (16,32,32,5)   81920
    float* O2 = out + 81920;    // rpn_reg NHWC (16,32,32,20)  327680
    float* O3 = out + 409600;   // roi_boxes (16,300,4)        19200
    float* O4 = out + 428800;   // cls (16,300,4)              19200
    float* O5 = out + 448000;   // reg (16,300,16)             76800
    float* O6 = out + 524800;   // dec_b (16,100,4)            6400
    float* O7 = out + 531200;   // dec_l (16,100)              1600
    float* O8 = out + 532800;   // dec_s (16,100)              1600

    char* wsb = (char*)d_ws;
    size_t off = 0;
    auto alloc = [&](size_t bytes) -> char* {
        char* p = wsb + off;
        off = (off + bytes + 255) & ~(size_t)255;
        return p;
    };
    bf16_t* WC12 = (bf16_t*)alloc((size_t)36864 * 2);
    bf16_t* WC21 = (bf16_t*)alloc((size_t)73728 * 2);
    bf16_t* WC22 = (bf16_t*)alloc((size_t)147456 * 2);
    bf16_t* WC31 = (bf16_t*)alloc((size_t)294912 * 2);
    bf16_t* WC32 = (bf16_t*)alloc((size_t)589824 * 2);
    bf16_t* WC33 = (bf16_t*)alloc((size_t)589824 * 2);
    bf16_t* WRPN = (bf16_t*)alloc((size_t)589824 * 2);
    bf16_t* WFC1 = (bf16_t*)alloc((size_t)4227072 * 2);
    bf16_t* WFC2 = (bf16_t*)alloc((size_t)524288 * 2);
    bf16_t* ACTA = (bf16_t*)alloc((size_t)16777216 * 2);
    bf16_t* ACTB = (bf16_t*)alloc((size_t)16777216 * 2);
    bf16_t* ACTC = (bf16_t*)alloc((size_t)4194304 * 2);
    float*  FMF  = (float*)alloc((size_t)4194304 * 4);
    bf16_t* FMB  = (bf16_t*)alloc((size_t)4194304 * 2);
    bf16_t* RPNA = (bf16_t*)alloc((size_t)4194304 * 2);
    float*  SCORES = (float*)alloc((size_t)81920 * 4);
    float*  BOXES  = (float*)alloc((size_t)327680 * 4);
    float*  TOPS   = (float*)alloc((size_t)16000 * 4);
    float*  TOPB   = (float*)alloc((size_t)64000 * 4);
    float*  ROIS   = (float*)alloc((size_t)4800 * 4);
    bf16_t* POOLED = (bf16_t*)alloc((size_t)19814400 * 2);
    bf16_t* FC1O   = (bf16_t*)alloc((size_t)4915200 * 2);
    bf16_t* FC2O   = (bf16_t*)alloc((size_t)2457600 * 2);
    float*  PROBS  = (float*)alloc((size_t)19200 * 4);
    int*    AMAX   = (int*)alloc((size_t)4800 * 4);
    float*  CB     = (float*)alloc((size_t)76800 * 4);
    float*  CS     = (float*)alloc((size_t)19200 * 4);

    // --- weight conversion into fragment-ready blocked bf16 ---
    convertB_conv_kernel<<<(576 * 64 + 255) / 256, 256, 0, stream>>>(b1w1, WC12, 64, 64);
    convertB_conv_kernel<<<(576 * 128 + 255) / 256, 256, 0, stream>>>(b2w0, WC21, 64, 128);
    convertB_conv_kernel<<<(1152 * 128 + 255) / 256, 256, 0, stream>>>(b2w1, WC22, 128, 128);
    convertB_conv_kernel<<<(1152 * 256 + 255) / 256, 256, 0, stream>>>(b3w0, WC31, 128, 256);
    convertB_conv_kernel<<<(2304 * 256 + 255) / 256, 256, 0, stream>>>(b3w1, WC32, 256, 256);
    convertB_conv_kernel<<<(2304 * 256 + 255) / 256, 256, 0, stream>>>(b3w2, WC33, 256, 256);
    convertB_conv_kernel<<<(2304 * 256 + 255) / 256, 256, 0, stream>>>(rpnw, WRPN, 256, 256);
    convertB_fc_kernel<<<(4128 * 1024 + 255) / 256, 256, 0, stream>>>(fc1w, WFC1, 4098, 4128, 1024);
    convertB_fc_kernel<<<(1024 * 512 + 255) / 256, 256, 0, stream>>>(fc2w, WFC2, 1024, 1024, 512);

    // --- backbone (all activations NHWC) ---
    conv1_kernel<<<65536, 256, 0, stream>>>(x, b1w0, b1b0, ACTA);
    conv3x3_wmma_kernel<<<dim3(2048, 1), 256, 0, stream>>>(ACTA, WC12, b1b1, ACTB, 64, 256, 64, 64, 0);
    maxpool_kernel<<<16384, 256, 0, stream>>>(ACTB, ACTC, nullptr, 64, 256, 64, 2, 2);
    conv3x3_wmma_kernel<<<dim3(512, 2), 256, 0, stream>>>(ACTC, WC21, b2b0, ACTA, 64, 128, 32, 128, 0);
    conv3x3_wmma_kernel<<<dim3(512, 2), 256, 0, stream>>>(ACTA, WC22, b2b1, ACTB, 128, 128, 32, 128, 0);
    maxpool_kernel<<<16384, 256, 0, stream>>>(ACTB, ACTC, nullptr, 128, 128, 32, 2, 1);
    conv3x3_wmma_kernel<<<dim3(256, 4), 256, 0, stream>>>(ACTC, WC31, b3b0, ACTA, 128, 64, 32, 256, 0);
    conv3x3_wmma_kernel<<<dim3(256, 4), 256, 0, stream>>>(ACTA, WC32, b3b1, ACTB, 256, 64, 32, 256, 0);
    conv3x3_wmma_kernel<<<dim3(256, 4), 256, 0, stream>>>(ACTB, WC33, b3b2, ACTA, 256, 64, 32, 256, 0);
    maxpool_kernel<<<16384, 256, 0, stream>>>(ACTA, FMB, FMF, 256, 64, 32, 2, 1);

    // --- RPN ---
    conv3x3_wmma_kernel<<<dim3(128, 4), 256, 0, stream>>>(FMB, WRPN, rpnb, RPNA, 256, 32, 32, 256, 1);
    rpn_heads_kernel<<<16384, 256, 0, stream>>>(RPNA, rpnclsw, rpnclsb, rpnregw, rpnregb, O1, O2);
    rpn_decode_kernel<<<(81920 + 255) / 256, 256, 0, stream>>>(O1, O2, SCORES, BOXES);
    rpn_topk_kernel<<<16, 1024, 0, stream>>>(SCORES, BOXES, TOPS, TOPB);
    rpn_nms_kernel<<<16, 1024, 0, stream>>>(TOPB, TOPS, O3, ROIS);

    // --- RoI head ---
    roi_pool_kernel<<<4800, 256, 0, stream>>>(FMF, O3, ROIS, POOLED);
    gemm_bf16_wmma_kernel<<<dim3(75, 8), 256, 0, stream>>>(POOLED, WFC1, fc1b, FC1O, 4800, 1024, 4128, 1);
    gemm_bf16_wmma_kernel<<<dim3(75, 4), 256, 0, stream>>>(FC1O, WFC2, fc2b, FC2O, 4800, 512, 1024, 1);
    fc_heads_kernel<<<(4800 * 20 + 255) / 256, 256, 0, stream>>>(FC2O, clsw, clsb, regw, regb, O4, O5);

    // --- decoder ---
    softmax_kernel<<<(4800 + 255) / 256, 256, 0, stream>>>(O4, PROBS, AMAX);
    perclass_kernel<<<64, 512, 0, stream>>>(PROBS, AMAX, O3, O5, CB, CS);
    final_topk_kernel<<<16, 1024, 0, stream>>>(CB, CS, O6, O7, O8);
}
